// Model3_36713380446231
// MI455X (gfx1250) — matrix operand order repaired and checked by reading, not compiled, not run on hardware
//
#include <hip/hip_runtime.h>
#include <hip/hip_bf16.h>

typedef __attribute__((ext_vector_type(16))) _Float16 v16h;
typedef __attribute__((ext_vector_type(8)))  _Float16 v8h;
typedef __attribute__((ext_vector_type(8)))  float    v8f;

// Problem constants (from reference)
constexpr int H   = 51;     // hidden
constexpr int G   = 204;    // 4*H gates
constexpr int GP  = 208;    // gates padded to multiple of 16
constexpr int KP  = 64;     // K (hidden) padded for 2x K=32 WMMA chunks
constexpr int B   = 1024;
constexpr int T   = 512;
constexpr int ROWS = 16;    // batch rows per workgroup (one WMMA M tile)
constexpr int THREADS = 128; // 4 waves

// ---- LDS layout (dynamic shared memory, byte offsets) ----
constexpr int OFF_W     = 0;                          // 6 x f16 [GP][KP] weight mats
constexpr int W_MAT     = GP * KP;                    // halfs per matrix
constexpr int OFF_GATES = OFF_W + 6 * W_MAT * 2;      // f32 [16][GP]
constexpr int OFF_H     = OFF_GATES + ROWS * GP * 4;  // 4 x f16 [16][KP]
constexpr int OFF_C     = OFF_H + 4 * ROWS * KP * 2;  // 4 x f32 [16][52]
constexpr int CPITCH    = 52;
constexpr int OFF_BIAS  = OFF_C + 4 * ROWS * CPITCH * 4; // 4 x f32 [GP]
constexpr int OFF_WX    = OFF_BIAS + 4 * GP * 4;         // 2 x f32 [GP] (wih1f/b)
constexpr int OFF_WL    = OFF_WX + 2 * GP * 4;           // 2 x f32 [52] (wlf/wlb)
constexpr int OFF_XS    = OFF_WL + 2 * 52 * 4;           // f32 [16]
constexpr int OFF_SC    = OFF_XS + 16 * 4;               // f32 scalars
constexpr int SMEM_BYTES = OFF_SC + 8 * 4;               // ~200 KB < 320 KB/WGP

// ---- Fast transcendentals on the CDNA5 TRANS unit (co-executes with WMMA) ----
#if defined(__has_builtin) && __has_builtin(__builtin_amdgcn_tanhf)
__device__ __forceinline__ float fast_tanh(float x) {   // single v_tanh_f32
    return __builtin_amdgcn_tanhf(x);
}
#else
__device__ __forceinline__ float fast_tanh(float x) {   // v_exp + v_rcp, no IEEE div
    const float e = __expf(2.f * x);
    return (e - 1.f) * __builtin_amdgcn_rcpf(e + 1.f);
}
#endif
__device__ __forceinline__ float sigm(float z) {        // 0.5 + 0.5*tanh(z/2)
    return __builtin_fmaf(0.5f, fast_tanh(0.5f * z), 0.5f);
}

// A fragment for V_WMMA_F32_16X16X32_F16:
// lane 0..15 : elements 0..15 -> K=kb+0..15
// lane 16..31: elements 0..15 -> K=kb+16..31
__device__ __forceinline__ v16h load_a(const _Float16* Arow, int kb, int kALo) {
    const v8h lo = *(const v8h*)(Arow + kb + kALo);
    const v8h hi = *(const v8h*)(Arow + kb + 8 + kALo);
    v16h r;
#pragma unroll
    for (int e = 0; e < 8; ++e) { r[e] = lo[e]; r[8 + e] = hi[e]; }
    return r;
}

// One GEMM: gates[16][GP] = A[16][KP] @ Wt[GP][KP]^T  (Wt stored n-major, k-contig)
__device__ __forceinline__ void gate_matmul1(const _Float16* A, const _Float16* Wt,
                                             float* gates, int tid) {
    const int lane = tid & 31, wv = tid >> 5;
    const int m0 = lane & 15;
    const int laneHi = lane >> 4;          // 0 or 1
    const int kALo = laneHi * 16;
    const int kBLo = laneHi * 16;
    for (int nt = wv; nt < GP / 16; nt += 4) {
        const int n = nt * 16 + m0;        // this lane's B column
        v8f acc = {};
#pragma unroll
        for (int kc = 0; kc < 2; ++kc) {
            const int kb = kc * 32;
            v16h a = load_a(A + m0 * KP, kb, kALo);
            v16h b = *(const v16h*)(Wt + n * KP + kb + kBLo);
            acc = __builtin_amdgcn_wmma_f32_16x16x32_f16(
                false, a, false, b, (short)0, acc, false, false);
        }
#pragma unroll
        for (int r = 0; r < 8; ++r)
            gates[(r + laneHi * 8) * GP + n] = acc[r];
    }
}

// Two fused GEMMs: gates = A1 @ W1^T + A2 @ W2^T   (layer-2 cells: ih + hh)
__device__ __forceinline__ void gate_matmul2(const _Float16* A1, const _Float16* W1,
                                             const _Float16* A2, const _Float16* W2,
                                             float* gates, int tid) {
    const int lane = tid & 31, wv = tid >> 5;
    const int m0 = lane & 15;
    const int laneHi = lane >> 4;
    const int kALo = laneHi * 16;
    const int kBLo = laneHi * 16;
    for (int nt = wv; nt < GP / 16; nt += 4) {
        const int n = nt * 16 + m0;
        v8f acc = {};
#pragma unroll
        for (int kc = 0; kc < 2; ++kc) {
            const int kb = kc * 32;
            v16h a = load_a(A1 + m0 * KP, kb, kALo);
            v16h b = *(const v16h*)(W1 + n * KP + kb + kBLo);
            acc = __builtin_amdgcn_wmma_f32_16x16x32_f16(
                false, a, false, b, (short)0, acc, false, false);
        }
#pragma unroll
        for (int kc = 0; kc < 2; ++kc) {
            const int kb = kc * 32;
            v16h a = load_a(A2 + m0 * KP, kb, kALo);
            v16h b = *(const v16h*)(W2 + n * KP + kb + kBLo);
            acc = __builtin_amdgcn_wmma_f32_16x16x32_f16(
                false, a, false, b, (short)0, acc, false, false);
        }
#pragma unroll
        for (int r = 0; r < 8; ++r)
            gates[(r + laneHi * 8) * GP + n] = acc[r];
    }
}

// Elementwise LSTM cell update. wx != nullptr -> layer-1 cell (adds x * wih).
__device__ __forceinline__ void gate_elem(const float* gates, const float* bias,
                                          const float* wx, const float* xs,
                                          float* c, _Float16* h16, int tid) {
    for (int idx = tid; idx < ROWS * H; idx += THREADS) {
        const int m = idx / H, j = idx - m * H;
        float gi = gates[m * GP + j]           + bias[j];
        float gf = gates[m * GP + H + j]       + bias[H + j];
        float gg = gates[m * GP + 2 * H + j]   + bias[2 * H + j];
        float go = gates[m * GP + 3 * H + j]   + bias[3 * H + j];
        if (wx) {
            const float x = xs[m];
            gi = __builtin_fmaf(x, wx[j],       gi);
            gf = __builtin_fmaf(x, wx[H + j],   gf);
            gg = __builtin_fmaf(x, wx[2*H + j], gg);
            go = __builtin_fmaf(x, wx[3*H + j], go);
        }
        const float cn = sigm(gf) * c[m * CPITCH + j] + sigm(gi) * fast_tanh(gg);
        c[m * CPITCH + j] = cn;
        h16[m * KP + j] = (_Float16)(sigm(go) * fast_tanh(cn));
    }
}

__global__ __launch_bounds__(THREADS)
void lstm_bidir_persistent(
    const float* __restrict__ input,
    const float* __restrict__ wih1f, const float* __restrict__ whh1f,
    const float* __restrict__ bih1f, const float* __restrict__ bhh1f,
    const float* __restrict__ wih2f, const float* __restrict__ whh2f,
    const float* __restrict__ bih2f, const float* __restrict__ bhh2f,
    const float* __restrict__ wih1b, const float* __restrict__ whh1b,
    const float* __restrict__ bih1b, const float* __restrict__ bhh1b,
    const float* __restrict__ wih2b, const float* __restrict__ whh2b,
    const float* __restrict__ bih2b, const float* __restrict__ bhh2b,
    const float* __restrict__ wlf, const float* __restrict__ blf,
    const float* __restrict__ wlb, const float* __restrict__ blb,
    const float* __restrict__ wl,  const float* __restrict__ bl,
    float* __restrict__ out) {

    extern __shared__ char smem[];
    _Float16* wmat  = (_Float16*)(smem + OFF_W);
    _Float16* w1f   = wmat + 0 * W_MAT;   // whh1f^T
    _Float16* w2fi  = wmat + 1 * W_MAT;   // wih2f^T
    _Float16* w2fh  = wmat + 2 * W_MAT;   // whh2f^T
    _Float16* w1b   = wmat + 3 * W_MAT;   // whh1b^T
    _Float16* w2bi  = wmat + 4 * W_MAT;   // wih2b^T
    _Float16* w2bh  = wmat + 5 * W_MAT;   // whh2b^T
    float*    gates = (float*)(smem + OFF_GATES);
    _Float16* hall  = (_Float16*)(smem + OFF_H);
    _Float16* h1f = hall, *h2f = hall + ROWS*KP, *h1b = hall + 2*ROWS*KP, *h2b = hall + 3*ROWS*KP;
    float*    call  = (float*)(smem + OFF_C);
    float*    c1f = call, *c2f = call + ROWS*CPITCH, *c1b = call + 2*ROWS*CPITCH, *c2b = call + 3*ROWS*CPITCH;
    float*    ball  = (float*)(smem + OFF_BIAS);
    float*    b1f = ball, *b2f = ball + GP, *b1b = ball + 2*GP, *b2b = ball + 3*GP;
    float*    wx1f = (float*)(smem + OFF_WX);
    float*    wx1b = wx1f + GP;
    float*    wlfS = (float*)(smem + OFF_WL);
    float*    wlbS = wlfS + 52;
    float*    xs   = (float*)(smem + OFF_XS);
    float*    sc   = (float*)(smem + OFF_SC);

    const int tid  = threadIdx.x;
    const int row0 = blockIdx.x * ROWS;

    // ---- Stage weights into LDS (transposed: Wt[n][k] = w[n*H + k]) ----
    {
        const float* srcs[6] = { whh1f, wih2f, whh2f, whh1b, wih2b, whh2b };
        for (int mtx = 0; mtx < 6; ++mtx) {
            const float* s = srcs[mtx];
            _Float16* d = wmat + mtx * W_MAT;
            for (int idx = tid; idx < W_MAT; idx += THREADS) {
                const int n = idx >> 6, k = idx & 63;
                const float v = (n < G && k < H) ? s[n * H + k] : 0.f;
                d[idx] = (_Float16)v;
            }
        }
        for (int idx = tid; idx < GP; idx += THREADS) {
            const bool ok = idx < G;
            b1f[idx] = ok ? (bih1f[idx] + bhh1f[idx]) : 0.f;
            b2f[idx] = ok ? (bih2f[idx] + bhh2f[idx]) : 0.f;
            b1b[idx] = ok ? (bih1b[idx] + bhh1b[idx]) : 0.f;
            b2b[idx] = ok ? (bih2b[idx] + bhh2b[idx]) : 0.f;
            wx1f[idx] = ok ? wih1f[idx] : 0.f;
            wx1b[idx] = ok ? wih1b[idx] : 0.f;
        }
        for (int idx = tid; idx < 52; idx += THREADS) {
            wlfS[idx] = (idx < H) ? wlf[idx] : 0.f;
            wlbS[idx] = (idx < H) ? wlb[idx] : 0.f;
        }
        for (int idx = tid; idx < 4 * ROWS * KP; idx += THREADS) hall[idx] = (_Float16)0.f;
        for (int idx = tid; idx < 4 * ROWS * CPITCH; idx += THREADS) call[idx] = 0.f;
        if (tid == 0) { sc[0] = blf[0]; sc[1] = blb[0]; sc[2] = wl[0]; sc[3] = wl[1]; sc[4] = bl[0]; }
    }
    __syncthreads();

    // ---- Sequential recurrence over T; everything stays in LDS/registers ----
    for (int t = 0; t < T; ++t) {
        if (tid < ROWS) {
            xs[tid] = input[(size_t)(row0 + tid) * T + t];
            if (t + 1 < T)  // gfx1250 global_prefetch_b8 for next step's x_t
                __builtin_prefetch(input + (size_t)(row0 + tid) * T + t + 1, 0, 1);
        }
        __syncthreads();

        // layer-1 forward cell
        gate_matmul1(h1f, w1f, gates, tid);
        __syncthreads();
        gate_elem(gates, b1f, wx1f, xs, c1f, h1f, tid);
        __syncthreads();

        // layer-2 forward cell (ih on h1f + hh on h2f, fused accumulation)
        gate_matmul2(h1f, w2fi, h2f, w2fh, gates, tid);
        __syncthreads();
        gate_elem(gates, b2f, nullptr, xs, c2f, h2f, tid);
        __syncthreads();

        // layer-1 backward cell (same x_t; flip is a no-op in the reference)
        gate_matmul1(h1b, w1b, gates, tid);
        __syncthreads();
        gate_elem(gates, b1b, wx1b, xs, c1b, h1b, tid);
        __syncthreads();

        // layer-2 backward cell
        gate_matmul2(h1b, w2bi, h2b, w2bh, gates, tid);
        __syncthreads();
        gate_elem(gates, b2b, nullptr, xs, c2b, h2b, tid);
        __syncthreads();

        // output linears: out = wl0*(h2f.wlf + blf) + wl1*(h2b.wlb + blb) + bl
        if (tid < ROWS) {
            float af = sc[0], ab = sc[1];
#pragma unroll 4
            for (int j = 0; j < H; ++j) {
                af = __builtin_fmaf((float)h2f[tid * KP + j], wlfS[j], af);
                ab = __builtin_fmaf((float)h2b[tid * KP + j], wlbS[j], ab);
            }
            out[(size_t)(row0 + tid) * T + t] = sc[2] * af + sc[3] * ab + sc[4];
        }
        __syncthreads();
    }
}

extern "C" void kernel_launch(void* const* d_in, const int* in_sizes, int n_in,
                              void* d_out, int out_size, void* d_ws, size_t ws_size,
                              hipStream_t stream) {
    (void)in_sizes; (void)n_in; (void)d_ws; (void)ws_size; (void)out_size;
    const float* input = (const float*)d_in[0];
    const float* wih1f = (const float*)d_in[1];
    const float* whh1f = (const float*)d_in[2];
    const float* bih1f = (const float*)d_in[3];
    const float* bhh1f = (const float*)d_in[4];
    const float* wih2f = (const float*)d_in[5];
    const float* whh2f = (const float*)d_in[6];
    const float* bih2f = (const float*)d_in[7];
    const float* bhh2f = (const float*)d_in[8];
    const float* wih1b = (const float*)d_in[9];
    const float* whh1b = (const float*)d_in[10];
    const float* bih1b = (const float*)d_in[11];
    const float* bhh1b = (const float*)d_in[12];
    const float* wih2b = (const float*)d_in[13];
    const float* whh2b = (const float*)d_in[14];
    const float* bih2b = (const float*)d_in[15];
    const float* bhh2b = (const float*)d_in[16];
    const float* wlf = (const float*)d_in[17];
    const float* blf = (const float*)d_in[18];
    const float* wlb = (const float*)d_in[19];
    const float* blb = (const float*)d_in[20];
    const float* wl  = (const float*)d_in[21];
    const float* bl  = (const float*)d_in[22];
    float* out = (float*)d_out;

    (void)hipFuncSetAttribute((const void*)lstm_bidir_persistent,
                              hipFuncAttributeMaxDynamicSharedMemorySize, SMEM_BYTES);

    dim3 grid(B / ROWS), block(THREADS);
    lstm_bidir_persistent<<<grid, block, SMEM_BYTES, stream>>>(
        input,
        wih1f, whh1f, bih1f, bhh1f, wih2f, whh2f, bih2f, bhh2f,
        wih1b, whh1b, bih1b, bhh1b, wih2b, whh2b, bih2b, bhh2b,
        wlf, blf, wlb, blb, wl, bl, out);
}
